// NGCF_3143916060680
// MI455X (gfx1250) — matrix operand order repaired and checked by reading, not compiled, not run on hardware
//
#include <hip/hip_runtime.h>
#include <math.h>

// ---------------------------------------------------------------------------
// NGCF forward for MI455X (gfx1250, wave32, WMMA).
// Bandwidth-bound workload: keep [N,64] state resident in L2 (192MB),
// use v_wmma_f32_16x16x32_f16 for the [N,64]x[64,64] dense layers with the
// bias + leaky_relu + L2-normalize epilogue fused into the same kernel.
// ---------------------------------------------------------------------------

typedef __attribute__((ext_vector_type(16))) _Float16 v16h;
typedef __attribute__((ext_vector_type(8)))  float    v8f;

#define N_USER  100000
#define N_ITEM  60000
#define N_NODES 160000      // N_USER + N_ITEM, divisible by 128
#define NNZ_E   2000000
#define BATCH   16384
#define D_EMB   64
#define LAYERS  3

// ---------------------------------------------------------------------------
// ego[n,:] = concat(user_emb, item_emb); also seed all_e[:, 0:64]
// one float4 per thread, N*16 threads
// ---------------------------------------------------------------------------
__global__ __launch_bounds__(256)
void ngcf_init(const float* __restrict__ user_emb,
               const float* __restrict__ item_emb,
               float* __restrict__ ego,
               float* __restrict__ all_e)
{
    int idx = blockIdx.x * 256 + threadIdx.x;          // [0, N*16)
    int n = idx >> 4, q = idx & 15;
    const float* src = (n < N_USER) ? (user_emb + (size_t)n * D_EMB)
                                    : (item_emb + (size_t)(n - N_USER) * D_EMB);
    float4 v = ((const float4*)src)[q];
    ((float4*)(ego   + (size_t)n * D_EMB))[q] = v;
    ((float4*)(all_e + (size_t)n * 256  ))[q] = v;     // layer-0 slab
}

// ---------------------------------------------------------------------------
// zero side buffer (graph-capturable; no hipMemset)
// ---------------------------------------------------------------------------
__global__ __launch_bounds__(256)
void ngcf_zero(float* __restrict__ side)
{
    int idx = blockIdx.x * 256 + threadIdx.x;          // [0, N*16)
    ((float4*)side)[idx] = make_float4(0.f, 0.f, 0.f, 0.f);
}

// ---------------------------------------------------------------------------
// convert W_gc/W_bi to f16, transposed to column-major [L][2][n][k] so every
// lane's WMMA B-fragment is one contiguous 32B run along K.
// ---------------------------------------------------------------------------
__global__ __launch_bounds__(256)
void ngcf_wconv(const float* __restrict__ Wgc,
                const float* __restrict__ Wbi,
                _Float16* __restrict__ wt)
{
    int idx = blockIdx.x * 256 + threadIdx.x;          // l*4096 + n*64 + k
    if (idx >= LAYERS * 64 * 64) return;
    int l = idx / 4096;
    int rem = idx - l * 4096;
    int n = rem >> 6, k = rem & 63;
    wt[(size_t)(l * 2 + 0) * 4096 + rem] = (_Float16)Wgc[(size_t)l * 4096 + k * 64 + n];
    wt[(size_t)(l * 2 + 1) * 4096 + rem] = (_Float16)Wbi[(size_t)l * 4096 + k * 64 + n];
}

// ---------------------------------------------------------------------------
// SpMM scatter: side[row] += val * ego[col].  16 lanes per edge, one float4
// column-chunk per lane -> coalesced 256B row gather (hits L2: ego is 41MB),
// global_atomic_add_f32 scatter (side also L2-resident).
// ---------------------------------------------------------------------------
__global__ __launch_bounds__(256)
void ngcf_spmm(const int*   __restrict__ rows,
               const int*   __restrict__ cols,
               const float* __restrict__ vals,
               const float* __restrict__ ego,
               float* __restrict__ side)
{
    int gid = blockIdx.x * 256 + threadIdx.x;          // [0, NNZ*16)
    int e = gid >> 4, q = gid & 15;
    int r = rows[e], c = cols[e];
    float v = vals[e];
    float4 x = ((const float4*)(ego + (size_t)c * D_EMB))[q];
    float* dst = side + (size_t)r * D_EMB + q * 4;
    atomicAdd(dst + 0, v * x.x);
    atomicAdd(dst + 1, v * x.y);
    atomicAdd(dst + 2, v * x.z);
    atomicAdd(dst + 3, v * x.w);
}

// ---------------------------------------------------------------------------
// Fused layer: ego_out = leaky_relu(side@Wgc + bgc + (ego*side)@Wbi + bbi)
//              all_e[:, 64*(layer+1):] = l2norm_rows(ego_out)
// 8 waves/block, each wave owns a 16-row tile and all 64 output columns.
// Both GEMMs accumulate into one f32 WMMA C fragment (K=64 -> 2 k-steps each
// -> 4 v_wmma per column tile, 16 per wave iteration).
// ---------------------------------------------------------------------------
__global__ __launch_bounds__(256)
void ngcf_layer_wmma(const float* __restrict__ ego,
                     const float* __restrict__ side,
                     const _Float16* __restrict__ wt,   // [L][2][64][64] (n,k)
                     const float* __restrict__ bgc,     // [L][64]
                     const float* __restrict__ bbi,     // [L][64]
                     float* __restrict__ ego_out,       // [N,64] unnormalized
                     float* __restrict__ all_e,         // [N,256]
                     int layer)
{
    __shared__ float tile[8][16 * 64];
    __shared__ float scl[8][16];

    const int w    = threadIdx.x >> 5;
    const int lane = threadIdx.x & 31;
    const int rl   = lane & 15;       // row within tile (A) / column (B,C)
    const int kh   = lane >> 4;       // lane-half selects K sub-block
    const int row_base = (blockIdx.x * 8 + w) * 16;   // N divides exactly

    // ---- A fragments (f16): side, and ego*side, K-steps 0/1 --------------
    // A k order used here (B matches): lane(M=rl), elems 0..15 -> K=ks*32+kh*16+e
    const float* sp = side + (size_t)(row_base + rl) * D_EMB;
    const float* ep = ego  + (size_t)(row_base + rl) * D_EMB;
    v16h a_s[2], a_p[2];
#pragma unroll
    for (int ks = 0; ks < 2; ++ks) {
        const int o0 = ks * 32 + kh * 16;
        const int o1 = o0 + 8;
        float4 s0 = *(const float4*)(sp + o0);
        float4 s1 = *(const float4*)(sp + o0 + 4);
        float4 s2 = *(const float4*)(sp + o1);
        float4 s3 = *(const float4*)(sp + o1 + 4);
        float4 e0 = *(const float4*)(ep + o0);
        float4 e1 = *(const float4*)(ep + o0 + 4);
        float4 e2 = *(const float4*)(ep + o1);
        float4 e3 = *(const float4*)(ep + o1 + 4);
        float sbuf[16] = { s0.x,s0.y,s0.z,s0.w, s1.x,s1.y,s1.z,s1.w,
                           s2.x,s2.y,s2.z,s2.w, s3.x,s3.y,s3.z,s3.w };
        float ebuf[16] = { e0.x,e0.y,e0.z,e0.w, e1.x,e1.y,e1.z,e1.w,
                           e2.x,e2.y,e2.z,e2.w, e3.x,e3.y,e3.z,e3.w };
#pragma unroll
        for (int e = 0; e < 16; ++e) {
            a_s[ks][e] = (_Float16)sbuf[e];
            a_p[ks][e] = (_Float16)(sbuf[e] * ebuf[e]);
        }
    }

    const _Float16* wgc = wt + (size_t)(layer * 2 + 0) * 64 * 64;
    const _Float16* wbi = wt + (size_t)(layer * 2 + 1) * 64 * 64;

#pragma unroll
    for (int nt = 0; nt < 4; ++nt) {
        const int ncol = nt * 16 + rl;
        // B layout: lane(N=rl), elems 0..15 -> K = ks*32 + kh*16 + e
        // (contiguous 32B in col-major wt)
        v16h bg0 = *(const v16h*)(wgc + (size_t)ncol * 64 +  0 + kh * 16);
        v16h bg1 = *(const v16h*)(wgc + (size_t)ncol * 64 + 32 + kh * 16);
        v16h bb0 = *(const v16h*)(wbi + (size_t)ncol * 64 +  0 + kh * 16);
        v16h bb1 = *(const v16h*)(wbi + (size_t)ncol * 64 + 32 + kh * 16);

        float bias = bgc[layer * 64 + ncol] + bbi[layer * 64 + ncol];
        v8f c;
#pragma unroll
        for (int r = 0; r < 8; ++r) c[r] = bias;

        c = __builtin_amdgcn_wmma_f32_16x16x32_f16(false, a_s[0], false, bg0,
                                                   (short)0, c, false, false);
        c = __builtin_amdgcn_wmma_f32_16x16x32_f16(false, a_s[1], false, bg1,
                                                   (short)0, c, false, false);
        c = __builtin_amdgcn_wmma_f32_16x16x32_f16(false, a_p[0], false, bb0,
                                                   (short)0, c, false, false);
        c = __builtin_amdgcn_wmma_f32_16x16x32_f16(false, a_p[1], false, bb1,
                                                   (short)0, c, false, false);

        // epilogue: leaky_relu(0.2), write unnormalized ego_out + LDS tile
        // C layout: VGPR r -> row r + 8*kh, column = rl (+ nt*16)
#pragma unroll
        for (int r = 0; r < 8; ++r) {
            float x = c[r];
            x = (x >= 0.0f) ? x : 0.2f * x;
            int mrow = r + kh * 8;
            ego_out[(size_t)(row_base + mrow) * D_EMB + ncol] = x;
            tile[w][mrow * 64 + ncol] = x;
        }
    }

    __syncthreads();
    if (lane < 16) {                       // one row per lane: sum of squares
        const float* t = &tile[w][lane * 64];
        float s = 0.f;
#pragma unroll
        for (int d = 0; d < 64; ++d) s += t[d] * t[d];
        scl[w][lane] = 1.0f / fmaxf(sqrtf(s), 1e-12f);
    }
    __syncthreads();

    float* dst = all_e + (size_t)row_base * 256 + (size_t)64 * (layer + 1);
#pragma unroll
    for (int i = 0; i < 32; ++i) {         // 16x64 tile, 32 elems/lane
        int idx = lane + i * 32;
        int m = idx >> 6, d = idx & 63;
        dst[(size_t)m * 256 + d] = tile[w][m * 64 + d] * scl[w][m];
    }
}

// ---------------------------------------------------------------------------
// predict: one wave per sample; 8 floats/lane over the 256-wide concat,
// wave32 shfl_xor reduction, sigmoid.
// ---------------------------------------------------------------------------
__global__ __launch_bounds__(256)
void ngcf_predict(const int*   __restrict__ users,
                  const int*   __restrict__ items,
                  const float* __restrict__ all_e,
                  const float* __restrict__ pred_W,   // [256]
                  const float* __restrict__ pred_b,   // [1]
                  float* __restrict__ out)
{
    int tid  = blockIdx.x * 256 + threadIdx.x;
    int b    = tid >> 5;                   // BATCH divides exactly
    int lane = tid & 31;
    const float* up = all_e + (size_t)users[b] * 256;
    const float* ip = all_e + (size_t)(N_USER + items[b]) * 256;
    float p = 0.f;
#pragma unroll
    for (int j = 0; j < 8; ++j) {
        int d = lane * 8 + j;
        p += up[d] * ip[d] * pred_W[d];
    }
#pragma unroll
    for (int m = 16; m >= 1; m >>= 1) p += __shfl_xor(p, m, 32);
    if (lane == 0) {
        float z = p + pred_b[0];
        out[b] = 1.0f / (1.0f + expf(-z));
    }
}

// ---------------------------------------------------------------------------
extern "C" void kernel_launch(void* const* d_in, const int* in_sizes, int n_in,
                              void* d_out, int out_size, void* d_ws, size_t ws_size,
                              hipStream_t stream)
{
    const int*   users    = (const int*)  d_in[0];
    const int*   items    = (const int*)  d_in[1];
    const int*   adj_rows = (const int*)  d_in[2];
    const int*   adj_cols = (const int*)  d_in[3];
    const float* adj_vals = (const float*)d_in[4];
    const float* user_emb = (const float*)d_in[5];
    const float* item_emb = (const float*)d_in[6];
    const float* W_gc     = (const float*)d_in[7];
    const float* b_gc     = (const float*)d_in[8];
    const float* W_bi     = (const float*)d_in[9];
    const float* b_bi     = (const float*)d_in[10];
    const float* pred_W   = (const float*)d_in[11];
    const float* pred_b   = (const float*)d_in[12];
    float* out = (float*)d_out;

    // workspace carve-out (~287 MB): egoA | egoB | side | all_e | f16 weights
    char* ws = (char*)d_ws;
    const size_t egoBytes = (size_t)N_NODES * D_EMB * sizeof(float);   // 41MB
    float*    egoA  = (float*)(ws);
    float*    egoB  = (float*)(ws + egoBytes);
    float*    side  = (float*)(ws + 2 * egoBytes);
    float*    all_e = (float*)(ws + 3 * egoBytes);                     // 164MB
    _Float16* wt    = (_Float16*)(ws + 3 * egoBytes +
                                  (size_t)N_NODES * 256 * sizeof(float));

    const int vecBlocks  = (N_NODES * 16) / 256;   // 10000
    const int spmmBlocks = (NNZ_E * 16) / 256;     // 125000
    const int gemmBlocks = N_NODES / 128;          // 1250
    const int predBlocks = BATCH / 8;              // 2048

    ngcf_init<<<vecBlocks, 256, 0, stream>>>(user_emb, item_emb, egoA, all_e);
    ngcf_wconv<<<(LAYERS * 64 * 64 + 255) / 256, 256, 0, stream>>>(W_gc, W_bi, wt);

    float* ego_in  = egoA;
    float* ego_out = egoB;
    for (int k = 0; k < LAYERS; ++k) {
        ngcf_zero<<<vecBlocks, 256, 0, stream>>>(side);
        ngcf_spmm<<<spmmBlocks, 256, 0, stream>>>(adj_rows, adj_cols, adj_vals,
                                                  ego_in, side);
        ngcf_layer_wmma<<<gemmBlocks, 256, 0, stream>>>(ego_in, side, wt,
                                                        b_gc, b_bi,
                                                        ego_out, all_e, k);
        float* t = ego_in; ego_in = ego_out; ego_out = t;
    }

    ngcf_predict<<<predBlocks, 256, 0, stream>>>(users, items, all_e,
                                                 pred_W, pred_b, out);
}
